// MultiHeadAttention_19731079758058
// MI455X (gfx1250) — matrix operand order repaired and checked by reading, not compiled, not run on hardware
//
#include <hip/hip_runtime.h>
#include <hip/hip_bf16.h>

typedef _Float16 h16;
typedef _Float16 v16h __attribute__((ext_vector_type(16)));
typedef _Float16 v8h  __attribute__((ext_vector_type(8)));
typedef float    v8f  __attribute__((ext_vector_type(8)));
typedef unsigned int       u32x4 __attribute__((ext_vector_type(4)));
typedef int                i32x8 __attribute__((ext_vector_type(8)));
typedef int                i32x4 __attribute__((ext_vector_type(4)));

#define NB   2
#define SEQ  2048
#define DIM  1024
#define NH   16
#define DKV  64
#define BSR  (NB * SEQ)   // 4096 rows

#define WMMA_F16F32(A, B, C) \
  __builtin_amdgcn_wmma_f32_16x16x32_f16(false, (A), false, (B), (short)0, (C), false, false)

// ---- Fragment loaders (CDNA5 ISA 7.12.2 VGPR layouts, wave32) ----
__device__ __forceinline__ v16h load_a_frag(const h16* base, int ld, int lane) {
  union { v16h v; v8h h[2]; } u;
  const h16* p = base + (size_t)(lane & 15) * ld + ((lane >> 4) << 3);
  u.h[0] = *(const v8h*)(p);
  u.h[1] = *(const v8h*)(p + 16);
  return u.v;
}
__device__ __forceinline__ v16h load_b_frag(const h16* base, int ld, int lane) {
  union { v16h v; v8h h[2]; } u;
  const h16* p = base + (size_t)(lane & 15) * ld + ((lane >> 4) << 3);
  u.h[0] = *(const v8h*)(p);
  u.h[1] = *(const v8h*)(p + 16);
  return u.v;
}

__device__ __forceinline__ float redmax16(float x) {
  x = fmaxf(x, __shfl_xor(x, 1));
  x = fmaxf(x, __shfl_xor(x, 2));
  x = fmaxf(x, __shfl_xor(x, 4));
  x = fmaxf(x, __shfl_xor(x, 8));
  return x;
}
__device__ __forceinline__ float redsum16(float x) {
  x += __shfl_xor(x, 1);
  x += __shfl_xor(x, 2);
  x += __shfl_xor(x, 4);
  x += __shfl_xor(x, 8);
  return x;
}

// ---- TDM: 2D tile load global -> LDS (ISA 08_async_tensor D# groups 0/1) ----
// data_size=2 bytes. tile_d0 elems along dim0 (contiguous), tile_d1 rows.
__device__ __forceinline__ void tdm_load_2d(const h16* gptr, unsigned lds_byte_off,
                                            unsigned tile_d0, unsigned tile_d1,
                                            unsigned tensor_d0, unsigned tensor_d1,
                                            unsigned long long stride0_elems) {
  const unsigned long long ga = (unsigned long long)(uintptr_t)gptr;
  u32x4 g0;
  g0[0] = 1u;                                   // count=1 (valid user descriptor)
  g0[1] = lds_byte_off;                         // lds_addr
  g0[2] = (unsigned)(ga & 0xFFFFFFFFu);         // global_addr[31:0]
  g0[3] = (unsigned)((ga >> 32) & 0x1FFFFFFu) | (2u << 30);  // global_addr[56:32], type=2
  i32x8 g1;
  g1[0] = (int)(1u << 16);                      // workgroup_mask=0, data_size=1 (2B)
  g1[1] = (int)((tensor_d0 & 0xFFFFu) << 16);   // tensor_dim0[15:0] at bits 63:48
  g1[2] = (int)(((tensor_d0 >> 16) & 0xFFFFu) | ((tensor_d1 & 0xFFFFu) << 16));
  g1[3] = (int)(((tensor_d1 >> 16) & 0xFFFFu) | ((tile_d0 & 0xFFFFu) << 16));
  g1[4] = (int)(tile_d1 & 0xFFFFu);             // tile_dim1; tile_dim2=0
  g1[5] = (int)(unsigned)(stride0_elems & 0xFFFFFFFFu);      // dim0_stride[31:0]
  g1[6] = (int)(unsigned)((stride0_elems >> 32) & 0xFFFFu);  // dim0_stride[47:32]
  g1[7] = 0;                                    // dim1_stride high = 0
  i32x4 z4 = {0, 0, 0, 0};
#if __clang_major__ >= 23
  i32x8 z8 = {0, 0, 0, 0, 0, 0, 0, 0};
  __builtin_amdgcn_tensor_load_to_lds(g0, g1, z4, z4, z8, 0);
#else
  __builtin_amdgcn_tensor_load_to_lds(g0, g1, z4, z4, 0);
#endif
}

// ---- f32 -> f16 conversion ----
__global__ void cvt_f32_to_f16(const float* __restrict__ in, h16* __restrict__ out, int n) {
  int i = blockIdx.x * blockDim.x + threadIdx.x;
  if (i < n) out[i] = (h16)in[i];
}

// ---- Projection GEMM: C[4096,1024] = X @ W^T + bias (32x64 tile, double-buffered) ----
// TRANS=0: store f16 as [B,H,S,DK].  TRANS=1: store f16 as [B,H,DK,S] (for V).
template <int TRANS>
__global__ void proj_gemm(const h16* __restrict__ X, const h16* __restrict__ W,
                          const float* __restrict__ bias, h16* __restrict__ out) {
  const int wave = (blockIdx.x * blockDim.x + threadIdx.x) >> 5;
  const int lane = threadIdx.x & 31;
  const int m0 = (wave & 127) * 32;
  const int n0 = (wave >> 7) * 64;
  const int lc = lane & 15, hl = lane >> 4;

  v8f acc[2][4];
  #pragma unroll
  for (int i = 0; i < 2; ++i)
    #pragma unroll
    for (int t = 0; t < 4; ++t) acc[i][t] = 0.0f;

  v16h a0 = load_a_frag(X + (size_t)m0 * DIM, DIM, lane);
  v16h a1 = load_a_frag(X + (size_t)(m0 + 16) * DIM, DIM, lane);
  v16h bf[4];
  #pragma unroll
  for (int t = 0; t < 4; ++t)
    bf[t] = load_b_frag(W + (size_t)(n0 + t * 16) * DIM, DIM, lane);

  for (int k = 32; k < DIM; k += 32) {
    v16h na0 = load_a_frag(X + (size_t)m0 * DIM + k, DIM, lane);
    v16h na1 = load_a_frag(X + (size_t)(m0 + 16) * DIM + k, DIM, lane);
    v16h nb[4];
    #pragma unroll
    for (int t = 0; t < 4; ++t)
      nb[t] = load_b_frag(W + (size_t)(n0 + t * 16) * DIM + k, DIM, lane);
    #pragma unroll
    for (int t = 0; t < 4; ++t) {
      acc[0][t] = WMMA_F16F32(a0, bf[t], acc[0][t]);
      acc[1][t] = WMMA_F16F32(a1, bf[t], acc[1][t]);
    }
    a0 = na0; a1 = na1;
    #pragma unroll
    for (int t = 0; t < 4; ++t) bf[t] = nb[t];
  }
  #pragma unroll
  for (int t = 0; t < 4; ++t) {
    acc[0][t] = WMMA_F16F32(a0, bf[t], acc[0][t]);
    acc[1][t] = WMMA_F16F32(a1, bf[t], acc[1][t]);
  }

  #pragma unroll
  for (int t = 0; t < 4; ++t) {
    const int n = n0 + t * 16 + lc;
    const float bn = bias[n];
    const int h = n >> 6, dk = n & 63;
    #pragma unroll
    for (int i = 0; i < 2; ++i) {
      #pragma unroll
      for (int v = 0; v < 8; ++v) {
        const int m = m0 + i * 16 + v + 8 * hl;
        const int b = m >> 11, s = m & 2047;
        const size_t bh = (size_t)(b * NH + h);
        const h16 val = (h16)(acc[i][t][v] + bn);
        if (TRANS)
          out[(bh * DKV + dk) * SEQ + s] = val;       // [B,H,DK,S]
        else
          out[(bh * SEQ + s) * DKV + dk] = val;       // [B,H,S,DK]
      }
    }
  }
}

// ---- Flash attention with TDM staging ----
// Block = 4 waves = 4 adjacent 16-query tiles of one (b,h).
// Per 32-key block, wave 0 TDM-stages K tile (32x64) and Vt tile (64x32) into LDS
// (double-buffered, synced via TENSORcnt + workgroup barrier); all 4 waves consume.
__global__ void flash_attn(const h16* __restrict__ Q, const h16* __restrict__ K,
                           const h16* __restrict__ Vt, const int* __restrict__ mask,
                           h16* __restrict__ Aout) {
  // single shared object -> LDS offset 0; layout (halves):
  // [0..2047]   buf0 K tile  (32 keys x 64 dk, row-major)
  // [2048..4095] buf0 V tile (64 dk x 32 keys, row-major)
  // [4096..6143] buf1 K tile
  // [6144..8191] buf1 V tile
  // [8192..10239] P tiles, 512 halves per wave
  __shared__ h16 ldsAll[2 * 2 * 2048 + 4 * 512];

  const int wib  = threadIdx.x >> 5;
  const int lane = threadIdx.x & 31;
  const int bh = blockIdx.x >> 5;                 // 32 blocks per (b,h)
  const int qchunk = blockIdx.x & 31;
  const int q0 = (qchunk * 4 + wib) * 16;
  const int b  = bh >> 4, h = bh & 15;
  const int lc = lane & 15, hl = lane >> 4;

  const h16* Qbh  = Q  + (size_t)bh * SEQ * DKV;
  const h16* Kbh  = K  + (size_t)bh * SEQ * DKV;
  const h16* Vtbh = Vt + (size_t)bh * DKV * SEQ;
  const int* Mb   = mask + (size_t)b * SEQ * SEQ;
  h16* P = ldsAll + 8192 + wib * 512;

  // loop-invariant Q fragments (16 rows x 64 dk)
  const v16h qa0 = load_a_frag(Qbh + (size_t)q0 * DKV +  0, DKV, lane);
  const v16h qa1 = load_a_frag(Qbh + (size_t)q0 * DKV + 32, DKV, lane);

  v8f o[4];
  #pragma unroll
  for (int t = 0; t < 4; ++t) o[t] = 0.0f;
  float mrow[8], lrow[8];
  #pragma unroll
  for (int v = 0; v < 8; ++v) { mrow[v] = -1e30f; lrow[v] = 0.0f; }

  // prologue: stage first K/V tiles into buffer 0
  if (wib == 0) {
    tdm_load_2d(Kbh,  0u,           DKV, 32, DKV, SEQ, DKV);   // K: 32 rows of 64
    tdm_load_2d(Vtbh, 2048u * 2u,   32, DKV, SEQ, DKV, SEQ);   // Vt: 64 rows of 32
    __builtin_amdgcn_s_wait_tensorcnt(0);
  }
  __syncthreads();

  for (int kt = 0; kt < SEQ; kt += 32) {
    const int cur = (kt >> 5) & 1;
    const h16* ldsK = ldsAll + cur * 4096;
    const h16* ldsV = ldsAll + cur * 4096 + 2048;

    // stage next block into the other buffer (its readers finished last iteration)
    if (wib == 0 && kt + 32 < SEQ) {
      const unsigned nxtOff = (unsigned)((cur ^ 1) * 4096) * 2u;
      tdm_load_2d(Kbh + (size_t)(kt + 32) * DKV, nxtOff,          DKV, 32, DKV, SEQ, DKV);
      tdm_load_2d(Vtbh + (kt + 32),              nxtOff + 4096u,  32, DKV, SEQ, DKV, SEQ);
    }

    // scores for 32 keys: two 16x16 C tiles, 2 WMMAs each (64-deep dk)
    v8f s0 = 0.0f, s1 = 0.0f;
    {
      v16h k0 = load_b_frag(ldsK, DKV, lane);                 // keys 0-15, dk 0-31
      v16h k1 = load_b_frag(ldsK + 32, DKV, lane);            // keys 0-15, dk 32-63
      v16h k2 = load_b_frag(ldsK + 16 * DKV, DKV, lane);      // keys 16-31, dk 0-31
      v16h k3 = load_b_frag(ldsK + 16 * DKV + 32, DKV, lane); // keys 16-31, dk 32-63
      s0 = WMMA_F16F32(qa0, k0, s0);
      s0 = WMMA_F16F32(qa1, k1, s0);
      s1 = WMMA_F16F32(qa0, k2, s1);
      s1 = WMMA_F16F32(qa1, k3, s1);
    }
    // V fragments from LDS (dk row fixed per lane, keys contiguous)
    v16h vf[4];
    #pragma unroll
    for (int t = 0; t < 4; ++t)
      vf[t] = load_b_frag(ldsV + (t * 16) * 32, 32, lane);

    float alpha[8], p0[8], p1[8];
    #pragma unroll
    for (int v = 0; v < 8; ++v) {
      const int qrow = q0 + v + 8 * hl;
      float x0 = s0[v] * 0.125f;   // 1/sqrt(64)
      float x1 = s1[v] * 0.125f;
      if (Mb[(size_t)qrow * SEQ + kt + lc] == 0)      x0 = -1e9f;
      if (Mb[(size_t)qrow * SEQ + kt + 16 + lc] == 0) x1 = -1e9f;
      const float tmax = redmax16(fmaxf(x0, x1));      // row max (16-lane reduce)
      const float mn = fmaxf(mrow[v], tmax);
      const float al = __expf(mrow[v] - mn);
      p0[v] = __expf(x0 - mn);
      p1[v] = __expf(x1 - mn);
      lrow[v] = lrow[v] * al + p0[v] + p1[v];          // per-lane partial sum (reduced once at end)
      mrow[v] = mn;
      alpha[v] = al;
    }
    #pragma unroll
    for (int t = 0; t < 4; ++t)
      #pragma unroll
      for (int v = 0; v < 8; ++v) o[t][v] = o[t][v] * alpha[v];

    // re-layout P (C-layout) -> A-layout through LDS
    #pragma unroll
    for (int v = 0; v < 8; ++v) {
      const int r = v + 8 * hl;
      P[r * 32 + lc]      = (h16)p0[v];
      P[r * 32 + 16 + lc] = (h16)p1[v];
    }
    const v16h pa = load_a_frag(P, 32, lane);

    // O += P(16x32) @ V(32x64)
    #pragma unroll
    for (int t = 0; t < 4; ++t) o[t] = WMMA_F16F32(pa, vf[t], o[t]);

    // drain TDM for next buffer, then block-wide handoff
    if (wib == 0 && kt + 32 < SEQ) __builtin_amdgcn_s_wait_tensorcnt(0);
    __syncthreads();
  }

  // finalize: one sum-reduce per row, normalize, write [B,S,D] f16
  #pragma unroll
  for (int v = 0; v < 8; ++v) lrow[v] = redsum16(lrow[v]);
  #pragma unroll
  for (int t = 0; t < 4; ++t) {
    const int col = h * DKV + t * 16 + lc;
    #pragma unroll
    for (int v = 0; v < 8; ++v) {
      const int qrow = q0 + v + 8 * hl;
      Aout[((size_t)(b * SEQ + qrow)) * DIM + col] = (h16)(o[t][v] / lrow[v]);
    }
  }
}

// ---- Output GEMM: out[4096,1024] f32 = A @ Wo^T + bo (32x64 tile, double-buffered) ----
__global__ void out_gemm(const h16* __restrict__ X, const h16* __restrict__ W,
                         const float* __restrict__ bias, float* __restrict__ out) {
  const int wave = (blockIdx.x * blockDim.x + threadIdx.x) >> 5;
  const int lane = threadIdx.x & 31;
  const int m0 = (wave & 127) * 32;
  const int n0 = (wave >> 7) * 64;
  const int lc = lane & 15, hl = lane >> 4;

  v8f acc[2][4];
  #pragma unroll
  for (int i = 0; i < 2; ++i)
    #pragma unroll
    for (int t = 0; t < 4; ++t) acc[i][t] = 0.0f;

  v16h a0 = load_a_frag(X + (size_t)m0 * DIM, DIM, lane);
  v16h a1 = load_a_frag(X + (size_t)(m0 + 16) * DIM, DIM, lane);
  v16h bf[4];
  #pragma unroll
  for (int t = 0; t < 4; ++t)
    bf[t] = load_b_frag(W + (size_t)(n0 + t * 16) * DIM, DIM, lane);

  for (int k = 32; k < DIM; k += 32) {
    v16h na0 = load_a_frag(X + (size_t)m0 * DIM + k, DIM, lane);
    v16h na1 = load_a_frag(X + (size_t)(m0 + 16) * DIM + k, DIM, lane);
    v16h nb[4];
    #pragma unroll
    for (int t = 0; t < 4; ++t)
      nb[t] = load_b_frag(W + (size_t)(n0 + t * 16) * DIM + k, DIM, lane);
    #pragma unroll
    for (int t = 0; t < 4; ++t) {
      acc[0][t] = WMMA_F16F32(a0, bf[t], acc[0][t]);
      acc[1][t] = WMMA_F16F32(a1, bf[t], acc[1][t]);
    }
    a0 = na0; a1 = na1;
    #pragma unroll
    for (int t = 0; t < 4; ++t) bf[t] = nb[t];
  }
  #pragma unroll
  for (int t = 0; t < 4; ++t) {
    acc[0][t] = WMMA_F16F32(a0, bf[t], acc[0][t]);
    acc[1][t] = WMMA_F16F32(a1, bf[t], acc[1][t]);
  }

  #pragma unroll
  for (int t = 0; t < 4; ++t) {
    const int n = n0 + t * 16 + lc;
    const float bn = bias[n];
    #pragma unroll
    for (int i = 0; i < 2; ++i)
      #pragma unroll
      for (int v = 0; v < 8; ++v) {
        const int m = m0 + i * 16 + v + 8 * hl;
        out[(size_t)m * DIM + n] = acc[i][t][v] + bn;
      }
  }
}

extern "C" void kernel_launch(void* const* d_in, const int* in_sizes, int n_in,
                              void* d_out, int out_size, void* d_ws, size_t ws_size,
                              hipStream_t stream) {
  const float* in_q = (const float*)d_in[0];
  const float* in_k = (const float*)d_in[1];
  const float* in_v = (const float*)d_in[2];
  const int*   mask = (const int*)d_in[3];
  const float* Wq = (const float*)d_in[4];  const float* bq = (const float*)d_in[5];
  const float* Wk = (const float*)d_in[6];  const float* bk = (const float*)d_in[7];
  const float* Wv = (const float*)d_in[8];  const float* bv = (const float*)d_in[9];
  const float* Wo = (const float*)d_in[10]; const float* bo = (const float*)d_in[11];

  const size_t actN = (size_t)BSR * DIM;   // 4M elements
  const size_t wN   = (size_t)DIM * DIM;   // 1M elements
  char* ws = (char*)d_ws;
  size_t off = 0;
  h16* Xq16 = (h16*)(ws + off); off += actN * 2;
  h16* Xk16 = (h16*)(ws + off); off += actN * 2;
  h16* Xv16 = (h16*)(ws + off); off += actN * 2;
  h16* Wq16 = (h16*)(ws + off); off += wN * 2;
  h16* Wk16 = (h16*)(ws + off); off += wN * 2;
  h16* Wv16 = (h16*)(ws + off); off += wN * 2;
  h16* Wo16 = (h16*)(ws + off); off += wN * 2;
  h16* Q16  = (h16*)(ws + off); off += actN * 2;
  h16* K16  = (h16*)(ws + off); off += actN * 2;
  h16* Vt16 = (h16*)(ws + off); off += actN * 2;
  h16* A16  = (h16*)(ws + off); off += actN * 2;

  const int cvtBlk = 256;
  cvt_f32_to_f16<<<(int)((actN + cvtBlk - 1) / cvtBlk), cvtBlk, 0, stream>>>(in_q, Xq16, (int)actN);
  cvt_f32_to_f16<<<(int)((actN + cvtBlk - 1) / cvtBlk), cvtBlk, 0, stream>>>(in_k, Xk16, (int)actN);
  cvt_f32_to_f16<<<(int)((actN + cvtBlk - 1) / cvtBlk), cvtBlk, 0, stream>>>(in_v, Xv16, (int)actN);
  cvt_f32_to_f16<<<(int)((wN + cvtBlk - 1) / cvtBlk), cvtBlk, 0, stream>>>(Wq, Wq16, (int)wN);
  cvt_f32_to_f16<<<(int)((wN + cvtBlk - 1) / cvtBlk), cvtBlk, 0, stream>>>(Wk, Wk16, (int)wN);
  cvt_f32_to_f16<<<(int)((wN + cvtBlk - 1) / cvtBlk), cvtBlk, 0, stream>>>(Wv, Wv16, (int)wN);
  cvt_f32_to_f16<<<(int)((wN + cvtBlk - 1) / cvtBlk), cvtBlk, 0, stream>>>(Wo, Wo16, (int)wN);

  // Projections: 2048 waves (32x64 tiles) -> 512 blocks of 128 threads
  proj_gemm<0><<<512, 128, 0, stream>>>(Xq16, Wq16, bq, Q16);
  proj_gemm<0><<<512, 128, 0, stream>>>(Xk16, Wk16, bk, K16);
  proj_gemm<1><<<512, 128, 0, stream>>>(Xv16, Wv16, bv, Vt16);

  // Attention: 1024 blocks x 4 waves (4 q-tiles per block share TDM-staged K/V)
  flash_attn<<<1024, 128, 0, stream>>>(Q16, K16, Vt16, mask, A16);

  // Output projection
  out_gemm<<<512, 128, 0, stream>>>(A16, Wo16, bo, (float*)d_out);
}
